// GCN_8220567405097
// MI455X (gfx1250) — matrix operand order repaired and checked by reading, not compiled, not run on hardware
//
#include <hip/hip_runtime.h>

// ---------------------------------------------------------------------------
// GCN (4-layer, eval mode) for MI455X / gfx1250, wave32.
//   layer: hlin = Xb(bf16) @ Wt(bf16)  [v_wmma_f32_16x16x32_bf16, f32 accum]
//          hagg[dst] += hlin[src] * dinv[src]*dinv[dst]   (f32 atomics, L2)
//          next Xb   = bf16(relu(bn(hagg + b)))           (fused epilogue)
// GEMM: templated on O so strides are immediates; per wave, B fragments
// (weight tile) loaded once and reused across 5 row tiles (20 WMMAs/wave);
// all 4 A fragments of a tile issued as one load clause before the WMMAs.
// ---------------------------------------------------------------------------

#define NN   50000
#define EPSV 1e-5f
#define RPW  5            // row tiles per wave (3125 = 5^5, divides evenly)
#define KK   128          // inner dimension (all layers)

typedef __attribute__((ext_vector_type(16))) __bf16        v16bf;
typedef __attribute__((ext_vector_type(8)))  float         v8f;
typedef __attribute__((ext_vector_type(8)))  unsigned int  v8u;

__device__ __forceinline__ unsigned short f32_to_bf16(float f) {
    unsigned int u = __builtin_bit_cast(unsigned int, f);
    u += 0x7FFFu + ((u >> 16) & 1u);               // round-to-nearest-even
    return (unsigned short)(u >> 16);
}

// ---------------- prep kernels ----------------
__global__ void k_fill1(float* __restrict__ p, int n) {
    int i = blockIdx.x * blockDim.x + threadIdx.x;
    if (i < n) p[i] = 1.0f;                         // self-loop contributes 1
}

__global__ void k_degacc(const int* __restrict__ dst, float* __restrict__ deg, int E) {
    int i = blockIdx.x * blockDim.x + threadIdx.x;
    if (i < E) atomicAdd(&deg[dst[i]], 1.0f);
}

__global__ void k_rsqrt_inplace(float* __restrict__ p, int n) {
    int i = blockIdx.x * blockDim.x + threadIdx.x;
    if (i < n) p[i] = rsqrtf(p[i]);
}

__global__ void k_cvt_bf16(const float* __restrict__ in, unsigned short* __restrict__ out, int n) {
    int i = blockIdx.x * blockDim.x + threadIdx.x;
    if (i < n) out[i] = f32_to_bf16(in[i]);
}

// W [K,O] row-major f32  ->  Wt [O,K] row-major bf16 (B-fragment loads contiguous in K)
__global__ void k_cvtT_w(const float* __restrict__ W, unsigned short* __restrict__ Wt,
                         int K, int O) {
    int i = blockIdx.x * blockDim.x + threadIdx.x;
    if (i < K * O) {
        int k = i / O, o = i - k * O;
        Wt[(size_t)o * K + k] = f32_to_bf16(W[i]);
    }
}

// ---------------- WMMA GEMM: C[N,O] = A[N,KK](bf16) * Wt[O,KK](bf16)^T ----------------
// One wave handles RPW consecutive 16x16 row tiles of one column tile.
// B fragments for all 4 K-steps loaded once into registers; per tile, all 4
// A fragments issued as one clause, then 4 WMMAs, then 8 const-offset stores.
// Whole-wave guard keeps EXEC all-ones inside WMMA as the ISA requires.
template <int O>
__global__ void __launch_bounds__(256)
k_gemm_wmma_bf16(const unsigned short* __restrict__ Ab,
                 const unsigned short* __restrict__ Bt,
                 float* __restrict__ C,
                 int total_groups) {
    const int lane  = threadIdx.x;                   // 0..31
    const int group = blockIdx.x * 8 + threadIdx.y;  // wave-uniform
    if (group >= total_groups) return;
    constexpr int tpr = O >> 4;                      // col tiles
    const int rg  = group / tpr;                     // row-tile group (RPW tiles)
    const int ct  = group - rg * tpr;
    const int l15 = lane & 15;
    const int hg  = lane >> 4;                       // half-wave K group

    // B: 32x16 bf16 per step; lane n=l15, pairs at k/2 = ks*16 + hg*8 + {0..7}
    const unsigned int* Brow = (const unsigned int*)(Bt + (size_t)(ct * 16 + l15) * KK);
    v8u bu[4];
#pragma unroll
    for (int ks = 0; ks < 4; ++ks) {
        const int bb = ks * 16 + hg * 8;
#pragma unroll
        for (int j = 0; j < 8; ++j) bu[ks][j] = Brow[bb + j];
    }

    const int row0 = rg * (16 * RPW) + l15;
    const unsigned int* Arow = (const unsigned int*)(Ab + (size_t)row0 * KK);
    float* Crow = C + (size_t)(rg * (16 * RPW) + hg * 8) * O + (size_t)(ct * 16 + l15);

#pragma unroll
    for (int t = 0; t < RPW; ++t) {
        // A: 16x32 bf16 per step; lane m=l15, pairs at k/2 = ks*16 + hg*8 + {0..7}
        // Load ALL fragments for this tile first (single clause, staged waits).
        v8u au[4];
#pragma unroll
        for (int ks = 0; ks < 4; ++ks) {
            const int ab = ks * 16 + hg * 8;
#pragma unroll
            for (int i = 0; i < 4; ++i) {
                au[ks][i]     = Arow[ab + i];
                au[ks][i + 4] = Arow[ab + 4 + i];
            }
        }
        v8f acc = {};
#pragma unroll
        for (int ks = 0; ks < 4; ++ks) {
            acc = __builtin_amdgcn_wmma_f32_16x16x32_bf16(
                      /*neg_a=*/false, __builtin_bit_cast(v16bf, au[ks]),
                      /*neg_b=*/false, __builtin_bit_cast(v16bf, bu[ks]),
                      /*c_mod=*/(short)0, acc, /*reuse_a=*/false, /*reuse_b=*/false);
        }
        // C/D layout: VGPR r -> M = hg*8 + r, N = l15; row stride O*4 B (immediate)
#pragma unroll
        for (int r = 0; r < 8; ++r) Crow[(size_t)r * O] = acc[r];

        Arow += 16 * (KK / 2);   // next row tile (dword units)
        Crow += (size_t)16 * O;
    }
}

// ---------------- aggregation ----------------
// init with the self-loop term: hagg[i] = hlin[i] * dinv[i]^2
__global__ void k_agg_init(const float* __restrict__ hlin, const float* __restrict__ dinv,
                           float* __restrict__ hagg, int n, int O) {
    int i = blockIdx.x * blockDim.x + threadIdx.x;
    if (i < n * O) {
        float d = dinv[i / O];
        hagg[i] = hlin[i] * d * d;
    }
}

// per (edge, 4-channel) thread: hagg[dst] += hlin[src] * dinv[src]*dinv[dst]
__global__ void k_edge_agg(const int* __restrict__ src, const int* __restrict__ dst,
                           const float* __restrict__ dinv, const float* __restrict__ hlin,
                           float* __restrict__ hagg, int E, int O) {
    const int cg = O >> 2;
    long long gid = (long long)blockIdx.x * blockDim.x + threadIdx.x;
    int e = (int)(gid / cg);
    if (e >= E) return;
    int c4 = (int)(gid - (long long)e * cg) * 4;
    int s = src[e], d = dst[e];
    float nrm = dinv[s] * dinv[d];
    const float4 hv = *(const float4*)(hlin + (size_t)s * O + c4);
    float* outp = hagg + (size_t)d * O + c4;
    atomicAdd(outp + 0, hv.x * nrm);
    atomicAdd(outp + 1, hv.y * nrm);
    atomicAdd(outp + 2, hv.z * nrm);
    atomicAdd(outp + 3, hv.w * nrm);
}

// ---------------- epilogues ----------------
// relu(bn(hagg + bias)) -> bf16 input of next layer (fused conversion)
__global__ void k_bias_bn_relu_bf16(const float* __restrict__ hagg,
                                    const float* __restrict__ b,  const float* __restrict__ g,
                                    const float* __restrict__ be, const float* __restrict__ mu,
                                    const float* __restrict__ var,
                                    unsigned short* __restrict__ outb, int n, int O) {
    int i = blockIdx.x * blockDim.x + threadIdx.x;
    if (i >= n * O) return;
    int c = i % O;
    float v = (hagg[i] + b[c] - mu[c]) * rsqrtf(var[c] + EPSV) * g[c] + be[c];
    outb[i] = f32_to_bf16(fmaxf(v, 0.0f));
}

// final layer: hagg + bias -> d_out (f32)
__global__ void k_bias_out(const float* __restrict__ hagg, const float* __restrict__ b,
                           float* __restrict__ out, int n, int O) {
    int i = blockIdx.x * blockDim.x + threadIdx.x;
    if (i < n * O) out[i] = hagg[i] + b[i % O];
}

// ---------------------------------------------------------------------------
extern "C" void kernel_launch(void* const* d_in, const int* in_sizes, int n_in,
                              void* d_out, int out_size, void* d_ws, size_t ws_size,
                              hipStream_t stream) {
    const float* x   = (const float*)d_in[0];
    const int*   ei  = (const int*)d_in[1];
    const int    E   = in_sizes[1] / 2;
    const int*   src = ei;
    const int*   dst = ei + E;
    const float* W[4]  = {(const float*)d_in[2], (const float*)d_in[4],
                          (const float*)d_in[6], (const float*)d_in[8]};
    const float* bb[4] = {(const float*)d_in[3], (const float*)d_in[5],
                          (const float*)d_in[7], (const float*)d_in[9]};
    const float* g[3]  = {(const float*)d_in[10], (const float*)d_in[14], (const float*)d_in[18]};
    const float* be[3] = {(const float*)d_in[11], (const float*)d_in[15], (const float*)d_in[19]};
    const float* mu[3] = {(const float*)d_in[12], (const float*)d_in[16], (const float*)d_in[20]};
    const float* vv[3] = {(const float*)d_in[13], (const float*)d_in[17], (const float*)d_in[21]};

    // workspace carve-out (256B aligned)
    char*  ws  = (char*)d_ws;
    size_t off = 0;
    auto carve = [&](size_t bytes) -> void* {
        void* p = ws + off;
        off += (bytes + 255) & ~(size_t)255;
        return p;
    };
    float*          dinv = (float*)carve((size_t)NN * 4);
    unsigned short* Xb   = (unsigned short*)carve((size_t)NN * 128 * 2);
    float*          hlin = (float*)carve((size_t)NN * 128 * 4);
    float*          hagg = (float*)carve((size_t)NN * 128 * 4);
    unsigned short* Wt[4];
    for (int l = 0; l < 4; ++l) Wt[l] = (unsigned short*)carve((size_t)128 * 128 * 2);

    const int TB = 256;
    auto nb = [&](long long n) { return (int)((n + TB - 1) / TB); };

    // degree -> dinv
    k_fill1<<<nb(NN), TB, 0, stream>>>(dinv, NN);
    k_degacc<<<nb(E), TB, 0, stream>>>(dst, dinv, E);
    k_rsqrt_inplace<<<nb(NN), TB, 0, stream>>>(dinv, NN);

    // bf16 copies: x and transposed weights
    k_cvt_bf16<<<nb((long long)NN * 128), TB, 0, stream>>>(x, Xb, NN * 128);
    const int Odim[4] = {128, 128, 128, 64};
    for (int l = 0; l < 4; ++l)
        k_cvtT_w<<<nb((long long)KK * Odim[l]), TB, 0, stream>>>(W[l], Wt[l], KK, Odim[l]);

    for (int l = 0; l < 4; ++l) {
        const int O = Odim[l];

        // dense GEMM (WMMA bf16): one wave per (5 row tiles x 1 col tile)
        const int total_groups = (NN / (16 * RPW)) * (O / 16);   // 625 * tpr
        dim3 gblk((total_groups + 7) / 8);
        dim3 gthr(32, 8);
        if (O == 128)
            k_gemm_wmma_bf16<128><<<gblk, gthr, 0, stream>>>(Xb, Wt[l], hlin, total_groups);
        else
            k_gemm_wmma_bf16<64><<<gblk, gthr, 0, stream>>>(Xb, Wt[l], hlin, total_groups);

        // aggregation: self loop init + edge scatter
        k_agg_init<<<nb((long long)NN * O), TB, 0, stream>>>(hlin, dinv, hagg, NN, O);
        k_edge_agg<<<nb((long long)E * (O / 4)), TB, 0, stream>>>(src, dst, dinv, hlin, hagg, E, O);

        // epilogue
        if (l < 3) {
            k_bias_bn_relu_bf16<<<nb((long long)NN * O), TB, 0, stream>>>(
                hagg, bb[l], g[l], be[l], mu[l], vv[l], Xb, NN, O);
        } else {
            k_bias_out<<<nb((long long)NN * O), TB, 0, stream>>>(hagg, bb[l], (float*)d_out, NN, O);
        }
    }
}
